// EGNN_46308337386412
// MI455X (gfx1250) — compile-verified, matrix operand order checked by reading, not run on hardware
//
#include <hip/hip_runtime.h>
#include <hip/hip_bf16.h>

// ---------------------------------------------------------------------------
// EGNN forward for MI455X (gfx1250), wave32 + WMMA bf16 16x16x32.
// h kept as fp32 master + bf16 mirror (gathers are the bandwidth bottleneck:
// bf16 halves them and feeds v_wmma_f32_16x16x32_bf16 directly).
// All per-layer weight matrices are pre-transposed to [n][k] bf16 so that both
// A and B WMMA fragments are contiguous 32B-per-lane loads (same K-permutation
// used on both sides => contraction is exact regardless of HW K ordering).
// Weight tiles are brought into LDS with CDNA5 async copies
// (global_load_async_to_lds_b128, ASYNCcnt) instead of VGPR round-trips.
// ---------------------------------------------------------------------------

typedef __attribute__((ext_vector_type(16))) __bf16 v16bf;
typedef __attribute__((ext_vector_type(8)))  float  v8f;

static constexpr int HDIM   = 128;  // hidden width
static constexpr int NTILE  = 8;    // 8 x 16 = 128 output cols
static constexpr int KCH    = 4;    // 4 x 32 = 128 K
static constexpr int WAVES  = 8;    // waves per workgroup
static constexpr int NTHR   = 256;  // threads per workgroup

__device__ __forceinline__ v16bf ldfrag(const __bf16* p) {
  return *reinterpret_cast<const v16bf*>(p);  // 32B: two b128 loads
}

__device__ __forceinline__ v16bf cvtfrag(const float* p) {
  v16bf r;
#pragma unroll
  for (int j = 0; j < 16; ++j) r[j] = (__bf16)p[j];  // v_cvt_pk_bf16_f32
  return r;
}

__device__ __forceinline__ v8f wmma_bf16(v16bf a, v16bf b, v8f c) {
  return __builtin_amdgcn_wmma_f32_16x16x32_bf16(false, a, false, b, (short)0, c,
                                                 false, false);
}

__device__ __forceinline__ float silu_f(float x) {
  return x * __builtin_amdgcn_rcpf(1.0f + __expf(-x));
}

__device__ __forceinline__ void atomAddF32(float* p, float v) {
  asm volatile("global_atomic_add_f32 %0, %1, off" ::"v"(p), "v"(v) : "memory");
}

__device__ __forceinline__ void waitDS() {
  asm volatile("s_wait_dscnt 0x0" ::: "memory");
}

// CDNA5 async global->LDS copy, 16B per lane, tracked on ASYNCcnt.
// LDS operand is the wave-relative LDS byte address = low 32 bits of the
// generic pointer (ISA: LDS aperture maps addr[31:0] to LDS).
__device__ __forceinline__ void asyncLds16(void* lds, const void* g) {
  unsigned l32 = (unsigned)(unsigned long long)lds;
  asm volatile("global_load_async_to_lds_b128 %0, %1, off" ::"v"(l32), "v"(g)
               : "memory");
}

__device__ __forceinline__ void waitAsync() {
  asm volatile("s_wait_asynccnt 0x0" ::: "memory");
}

// ---------------------------------------------------------------------------
// Weight prep: fp32 [K][128] row-major -> bf16 [128][K] (n-major, contiguous K)
// ---------------------------------------------------------------------------
__global__ void prep_transpose(const float* __restrict__ src,
                               __bf16* __restrict__ dst, int K) {
  int i = blockIdx.x * blockDim.x + threadIdx.x;
  if (i >= K * HDIM) return;
  int k = i / HDIM, n = i % HDIM;
  dst[n * K + k] = (__bf16)src[i];
}

// ---------------------------------------------------------------------------
// Embedding: h = h0[N,64] @ embW + b ; also zero agg/dpos, copy pos.
// ---------------------------------------------------------------------------
__global__ void __launch_bounds__(NTHR) egnn_embed(
    const float* __restrict__ h0, const float* __restrict__ posIn,
    const __bf16* __restrict__ Wt, const float* __restrict__ bias,
    float* __restrict__ h, __bf16* __restrict__ hbf,
    float* __restrict__ agg, float* __restrict__ posc, float* __restrict__ dpos,
    int Nn) {
  extern __shared__ __align__(128) unsigned char smem[];
  __bf16* sW = (__bf16*)smem;                 // 128*64 bf16 = 16KB
  float* sb = (float*)(smem + 128 * 64 * 2);  // 128 f32
  int tid = threadIdx.x;
  {
    for (int i = tid; i < 1024; i += NTHR) asyncLds16(sW + i * 8, Wt + i * 8);
    for (int i = tid; i < 128; i += NTHR) sb[i] = bias[i];
    waitAsync();
  }
  __syncthreads();
  int wave = tid >> 5, lane = tid & 31, n16 = lane & 15, half = lane >> 4;
  int nTiles = Nn >> 4;
  for (int t = blockIdx.x * WAVES + wave; t < nTiles; t += gridDim.x * WAVES) {
    int row = t * 16 + n16;
    const float* ap = h0 + (size_t)row * 64 + half * 16;
    v16bf a[2];
    a[0] = cvtfrag(ap);
    a[1] = cvtfrag(ap + 32);
#pragma unroll
    for (int nt = 0; nt < NTILE; ++nt) {
      v8f c = {0, 0, 0, 0, 0, 0, 0, 0};
#pragma unroll
      for (int kc = 0; kc < 2; ++kc)
        c = wmma_bf16(a[kc], ldfrag(sW + (nt * 16 + n16) * 64 + kc * 32 + half * 16), c);
      int col = nt * 16 + n16;
      float bb = sb[col];
#pragma unroll
      for (int v = 0; v < 8; ++v) {
        int r = t * 16 + v + 8 * half;
        float x = c[v] + bb;
        size_t idx = (size_t)r * HDIM + col;
        h[idx] = x;
        hbf[idx] = (__bf16)x;
        agg[idx] = 0.0f;
      }
    }
    if (lane < 16) {
      int r = t * 16 + lane;
#pragma unroll
      for (int j = 0; j < 3; ++j) {
        posc[r * 3 + j] = posIn[r * 3 + j];
        dpos[r * 3 + j] = 0.0f;
      }
    }
  }
}

// ---------------------------------------------------------------------------
// Edge kernel: message MLP (2 layers) + scatter-add agg, coord MLP + scatter
// dpos.  One wave = 16 edges x 128 cols.  K=257 split into hi@W1a + hj@W1b +
// dist*w1c (fp32, exact).
// ---------------------------------------------------------------------------
__global__ void __launch_bounds__(NTHR) egnn_edge(
    const float* __restrict__ posc, const __bf16* __restrict__ hbf,
    const int* __restrict__ ei, int E_, int nEdges,
    const __bf16* __restrict__ W1a, const __bf16* __restrict__ W1b,
    const float* __restrict__ w1c, const float* __restrict__ b1,
    const __bf16* __restrict__ W2, const float* __restrict__ b2,
    const __bf16* __restrict__ C1, const float* __restrict__ cb1,
    const float* __restrict__ cw2, const float* __restrict__ cb2,
    float* __restrict__ agg, float* __restrict__ dpos) {
  extern __shared__ __align__(128) unsigned char smem[];
  __bf16* sW1a = (__bf16*)(smem + 0);
  __bf16* sW1b = (__bf16*)(smem + 32768);
  __bf16* sW2 = (__bf16*)(smem + 65536);
  __bf16* sC1 = (__bf16*)(smem + 98304);
  float* sw1c = (float*)(smem + 131072);
  float* sb1 = sw1c + 128;
  float* sb2 = sw1c + 256;
  float* scb1 = sw1c + 384;
  float* scw2 = sw1c + 512;
  __bf16* sStage = (__bf16*)(smem + 133632);  // 8 waves * 16*128 bf16
  float* sWsc = (float*)(smem + 166400);      // 8 waves * 16 f32
  int tid = threadIdx.x;
  {
    for (int i = tid; i < 2048; i += NTHR) {
      asyncLds16(sW1a + i * 8, W1a + i * 8);
      asyncLds16(sW1b + i * 8, W1b + i * 8);
      asyncLds16(sW2 + i * 8, W2 + i * 8);
      asyncLds16(sC1 + i * 8, C1 + i * 8);
    }
    for (int i = tid; i < 128; i += NTHR) {
      sw1c[i] = w1c[i]; sb1[i] = b1[i]; sb2[i] = b2[i];
      scb1[i] = cb1[i]; scw2[i] = cw2[i];
    }
    waitAsync();
  }
  __syncthreads();
  float cb2v = cb2[0];
  int wave = tid >> 5, lane = tid & 31, n16 = lane & 15, half = lane >> 4;
  __bf16* myStage = sStage + wave * (16 * HDIM);
  float* myW = sWsc + wave * 16;
  int nTiles = nEdges >> 4;
  int stride = gridDim.x * WAVES;
  for (int t = blockIdx.x * WAVES + wave; t < nTiles; t += stride) {
    int e = t * 16 + n16;
    // prefetch next tile's edge indices (global_prefetch_b8)
    __builtin_prefetch(ei + (e + stride * 16 < E_ ? e + stride * 16 : 0), 0, 1);
    int s_, d_;
    if (e < E_) {
      s_ = ei[e];
      d_ = ei[E_ + e];
    } else {
      s_ = e - E_;
      d_ = s_;
    }
    float rx = posc[s_ * 3 + 0] - posc[d_ * 3 + 0];
    float ry = posc[s_ * 3 + 1] - posc[d_ * 3 + 1];
    float rz = posc[s_ * 3 + 2] - posc[d_ * 3 + 2];
    float dist = sqrtf(rx * rx + ry * ry + rz * rz);
    const __bf16* hip = hbf + (size_t)d_ * HDIM + half * 16;
    const __bf16* hjp = hbf + (size_t)s_ * HDIM + half * 16;
    v16bf ahi[KCH], ahj[KCH];
#pragma unroll
    for (int kc = 0; kc < KCH; ++kc) {
      ahi[kc] = ldfrag(hip + kc * 32);
      ahj[kc] = ldfrag(hjp + kc * 32);
    }
    float dr[8];
    int ddr[8];
#pragma unroll
    for (int v = 0; v < 8; ++v) {
      int sl = v + 8 * half;                 // lane holding edge == row (v+8*half)
      dr[v] = __shfl(dist, sl, 32);
      ddr[v] = __shfl(d_, sl, 32);
    }
    // ---- message layer 1:  hi@W1a + hj@W1b + dist*w1c + b1, SiLU -> staging
#pragma unroll
    for (int nt = 0; nt < NTILE; ++nt) {
      v8f c = {0, 0, 0, 0, 0, 0, 0, 0};
#pragma unroll
      for (int kc = 0; kc < KCH; ++kc) {
        c = wmma_bf16(ahi[kc], ldfrag(sW1a + (nt * 16 + n16) * HDIM + kc * 32 + half * 16), c);
        c = wmma_bf16(ahj[kc], ldfrag(sW1b + (nt * 16 + n16) * HDIM + kc * 32 + half * 16), c);
      }
      int col = nt * 16 + n16;
      float wc = sw1c[col], bb = sb1[col];
#pragma unroll
      for (int v = 0; v < 8; ++v) {
        float x = silu_f(c[v] + dr[v] * wc + bb);
        myStage[(v + 8 * half) * HDIM + col] = (__bf16)x;
      }
    }
    waitDS();
    v16bf a2[KCH];
#pragma unroll
    for (int kc = 0; kc < KCH; ++kc)
      a2[kc] = ldfrag(myStage + n16 * HDIM + kc * 32 + half * 16);
    // ---- message layer 2 + scatter-add into agg[dst]
#pragma unroll
    for (int nt = 0; nt < NTILE; ++nt) {
      v8f c = {0, 0, 0, 0, 0, 0, 0, 0};
#pragma unroll
      for (int kc = 0; kc < KCH; ++kc)
        c = wmma_bf16(a2[kc], ldfrag(sW2 + (nt * 16 + n16) * HDIM + kc * 32 + half * 16), c);
      int col = nt * 16 + n16;
      float bb = sb2[col];
#pragma unroll
      for (int v = 0; v < 8; ++v) {
        float x = silu_f(c[v] + bb);
        atomAddF32(agg + (size_t)ddr[v] * HDIM + col, x);
      }
    }
    // ---- coord MLP: silu(hj@C1 + cb1) . cw2 + cb2  -> scalar per edge
    float p[8] = {0, 0, 0, 0, 0, 0, 0, 0};
#pragma unroll
    for (int nt = 0; nt < NTILE; ++nt) {
      v8f c = {0, 0, 0, 0, 0, 0, 0, 0};
#pragma unroll
      for (int kc = 0; kc < KCH; ++kc)
        c = wmma_bf16(ahj[kc], ldfrag(sC1 + (nt * 16 + n16) * HDIM + kc * 32 + half * 16), c);
      int col = nt * 16 + n16;
      float bb = scb1[col], wv = scw2[col];
#pragma unroll
      for (int v = 0; v < 8; ++v) p[v] += silu_f(c[v] + bb) * wv;
    }
#pragma unroll
    for (int v = 0; v < 8; ++v) {
      float x = p[v];
      x += __shfl_xor(x, 1, 32);
      x += __shfl_xor(x, 2, 32);
      x += __shfl_xor(x, 4, 32);
      x += __shfl_xor(x, 8, 32);             // reduce 16 lanes within half
      if (n16 == 0) myW[8 * half + v] = x;
    }
    waitDS();
    if (lane < 16) {
      float w = myW[lane] + cb2v;
      float cf = w * __builtin_amdgcn_rcpf(dist + 1e-8f);
      atomAddF32(dpos + (size_t)d_ * 3 + 0, cf * rx);
      atomAddF32(dpos + (size_t)d_ * 3 + 1, cf * ry);
      atomAddF32(dpos + (size_t)d_ * 3 + 2, cf * rz);
    }
  }
}

// ---------------------------------------------------------------------------
// Node update: h += silu([h,agg]@W1 + b1)@W2 + b2 ; pos += dpos; zero agg/dpos
// ---------------------------------------------------------------------------
__global__ void __launch_bounds__(NTHR) egnn_node(
    float* __restrict__ h, __bf16* __restrict__ hbf,
    float* __restrict__ agg, const __bf16* __restrict__ W1a,
    const __bf16* __restrict__ W1b, const float* __restrict__ b1,
    const __bf16* __restrict__ W2, const float* __restrict__ b2,
    float* __restrict__ posc, float* __restrict__ dpos, int Nn) {
  extern __shared__ __align__(128) unsigned char smem[];
  __bf16* sW1a = (__bf16*)smem;
  __bf16* sW1b = (__bf16*)(smem + 32768);
  __bf16* sW2 = (__bf16*)(smem + 65536);
  float* sb1 = (float*)(smem + 98304);
  float* sb2 = sb1 + 128;
  __bf16* sStage = (__bf16*)(smem + 99328);
  int tid = threadIdx.x;
  {
    for (int i = tid; i < 2048; i += NTHR) {
      asyncLds16(sW1a + i * 8, W1a + i * 8);
      asyncLds16(sW1b + i * 8, W1b + i * 8);
      asyncLds16(sW2 + i * 8, W2 + i * 8);
    }
    for (int i = tid; i < 128; i += NTHR) { sb1[i] = b1[i]; sb2[i] = b2[i]; }
    waitAsync();
  }
  __syncthreads();
  int wave = tid >> 5, lane = tid & 31, n16 = lane & 15, half = lane >> 4;
  __bf16* myStage = sStage + wave * (16 * HDIM);
  int nTiles = Nn >> 4;
  for (int t = blockIdx.x * WAVES + wave; t < nTiles; t += gridDim.x * WAVES) {
    int row = t * 16 + n16;
    const float* hp = h + (size_t)row * HDIM + half * 16;
    const float* gp = agg + (size_t)row * HDIM + half * 16;
    v16bf ah[KCH], ag[KCH];
#pragma unroll
    for (int kc = 0; kc < KCH; ++kc) {
      ah[kc] = cvtfrag(hp + kc * 32);
      ag[kc] = cvtfrag(gp + kc * 32);
    }
#pragma unroll
    for (int nt = 0; nt < NTILE; ++nt) {
      v8f c = {0, 0, 0, 0, 0, 0, 0, 0};
#pragma unroll
      for (int kc = 0; kc < KCH; ++kc) {
        c = wmma_bf16(ah[kc], ldfrag(sW1a + (nt * 16 + n16) * HDIM + kc * 32 + half * 16), c);
        c = wmma_bf16(ag[kc], ldfrag(sW1b + (nt * 16 + n16) * HDIM + kc * 32 + half * 16), c);
      }
      int col = nt * 16 + n16;
      float bb = sb1[col];
#pragma unroll
      for (int v = 0; v < 8; ++v)
        myStage[(v + 8 * half) * HDIM + col] = (__bf16)silu_f(c[v] + bb);
    }
    waitDS();
    v16bf a2[KCH];
#pragma unroll
    for (int kc = 0; kc < KCH; ++kc)
      a2[kc] = ldfrag(myStage + n16 * HDIM + kc * 32 + half * 16);
#pragma unroll
    for (int nt = 0; nt < NTILE; ++nt) {
      v8f c = {0, 0, 0, 0, 0, 0, 0, 0};
#pragma unroll
      for (int kc = 0; kc < KCH; ++kc)
        c = wmma_bf16(a2[kc], ldfrag(sW2 + (nt * 16 + n16) * HDIM + kc * 32 + half * 16), c);
      int col = nt * 16 + n16;
      float bb = sb2[col];
#pragma unroll
      for (int v = 0; v < 8; ++v) {
        int r = t * 16 + v + 8 * half;
        size_t idx = (size_t)r * HDIM + col;
        float nv = h[idx] + c[v] + bb;      // residual
        h[idx] = nv;
        hbf[idx] = (__bf16)nv;
        agg[idx] = 0.0f;                    // ready for next layer
      }
    }
    if (lane < 16) {
      int r = t * 16 + lane;
#pragma unroll
      for (int j = 0; j < 3; ++j) {
        float pp = posc[r * 3 + j] + dpos[r * 3 + j];
        posc[r * 3 + j] = pp;
        dpos[r * 3 + j] = 0.0f;
      }
    }
  }
}

// ---------------------------------------------------------------------------
// Output head: out = silu(h@oW1 + b1)@oW2 + b2 ; also emit final pos.
// ---------------------------------------------------------------------------
__global__ void __launch_bounds__(NTHR) egnn_out(
    const __bf16* __restrict__ hbf, const __bf16* __restrict__ W1,
    const float* __restrict__ b1, const __bf16* __restrict__ W2,
    const float* __restrict__ b2, const float* __restrict__ posc,
    float* __restrict__ outv, float* __restrict__ outp, int Nn) {
  extern __shared__ __align__(128) unsigned char smem[];
  __bf16* sW1 = (__bf16*)smem;
  __bf16* sW2 = (__bf16*)(smem + 32768);
  float* sb1 = (float*)(smem + 65536);
  float* sb2 = sb1 + 128;
  __bf16* sStage = (__bf16*)(smem + 66560);
  int tid = threadIdx.x;
  {
    for (int i = tid; i < 2048; i += NTHR) {
      asyncLds16(sW1 + i * 8, W1 + i * 8);
      asyncLds16(sW2 + i * 8, W2 + i * 8);
    }
    for (int i = tid; i < 128; i += NTHR) { sb1[i] = b1[i]; sb2[i] = b2[i]; }
    waitAsync();
  }
  __syncthreads();
  int wave = tid >> 5, lane = tid & 31, n16 = lane & 15, half = lane >> 4;
  __bf16* myStage = sStage + wave * (16 * HDIM);
  int nTiles = Nn >> 4;
  for (int t = blockIdx.x * WAVES + wave; t < nTiles; t += gridDim.x * WAVES) {
    int row = t * 16 + n16;
    v16bf a[KCH];
#pragma unroll
    for (int kc = 0; kc < KCH; ++kc)
      a[kc] = ldfrag(hbf + (size_t)row * HDIM + kc * 32 + half * 16);
#pragma unroll
    for (int nt = 0; nt < NTILE; ++nt) {
      v8f c = {0, 0, 0, 0, 0, 0, 0, 0};
#pragma unroll
      for (int kc = 0; kc < KCH; ++kc)
        c = wmma_bf16(a[kc], ldfrag(sW1 + (nt * 16 + n16) * HDIM + kc * 32 + half * 16), c);
      int col = nt * 16 + n16;
      float bb = sb1[col];
#pragma unroll
      for (int v = 0; v < 8; ++v)
        myStage[(v + 8 * half) * HDIM + col] = (__bf16)silu_f(c[v] + bb);
    }
    waitDS();
    v16bf a2[KCH];
#pragma unroll
    for (int kc = 0; kc < KCH; ++kc)
      a2[kc] = ldfrag(myStage + n16 * HDIM + kc * 32 + half * 16);
#pragma unroll
    for (int nt = 0; nt < NTILE; ++nt) {
      v8f c = {0, 0, 0, 0, 0, 0, 0, 0};
#pragma unroll
      for (int kc = 0; kc < KCH; ++kc)
        c = wmma_bf16(a2[kc], ldfrag(sW2 + (nt * 16 + n16) * HDIM + kc * 32 + half * 16), c);
      int col = nt * 16 + n16;
      float bb = sb2[col];
#pragma unroll
      for (int v = 0; v < 8; ++v) {
        int r = t * 16 + v + 8 * half;
        outv[(size_t)r * HDIM + col] = c[v] + bb;
      }
    }
    if (lane < 16) {
      int r = t * 16 + lane;
#pragma unroll
      for (int j = 0; j < 3; ++j) outp[r * 3 + j] = posc[r * 3 + j];
    }
  }
}

// ---------------------------------------------------------------------------
extern "C" void kernel_launch(void* const* d_in, const int* in_sizes, int n_in,
                              void* d_out, int out_size, void* d_ws, size_t ws_size,
                              hipStream_t stream) {
  const int N = 50000, E = 640000, L = 4;
  const int Etot = E + N;

  const float* h0 = (const float*)d_in[0];
  const float* posI = (const float*)d_in[1];
  const int* ei = (const int*)d_in[2];
  const float* embW = (const float*)d_in[3];
  const float* embB = (const float*)d_in[4];
  const float* msgW1 = (const float*)d_in[5];
  const float* msgB1 = (const float*)d_in[6];
  const float* msgW2 = (const float*)d_in[7];
  const float* msgB2 = (const float*)d_in[8];
  const float* coW1 = (const float*)d_in[9];
  const float* coB1 = (const float*)d_in[10];
  const float* coW2 = (const float*)d_in[11];
  const float* coB2 = (const float*)d_in[12];
  const float* noW1 = (const float*)d_in[13];
  const float* noB1 = (const float*)d_in[14];
  const float* noW2 = (const float*)d_in[15];
  const float* noB2 = (const float*)d_in[16];
  const float* oW1 = (const float*)d_in[17];
  const float* oB1 = (const float*)d_in[18];
  const float* oW2 = (const float*)d_in[19];
  const float* oB2 = (const float*)d_in[20];

  // ---- workspace carve (~67 MB)
  char* base = (char*)d_ws;
  auto take = [&](size_t bytes) -> char* {
    char* p = base;
    base += (bytes + 255) & ~(size_t)255;
    return p;
  };
  float* h = (float*)take((size_t)N * HDIM * 4);
  float* agg = (float*)take((size_t)N * HDIM * 4);
  __bf16* hbf = (__bf16*)take((size_t)N * HDIM * 2);
  float* posc = (float*)take((size_t)N * 3 * 4);
  float* dpos = (float*)take((size_t)N * 3 * 4);
  __bf16* wb = (__bf16*)take((size_t)499712 * 2);

  const size_t LSTR = 7 * 16384;
  __bf16* embT = wb;
  auto m1a = [&](int l) { return wb + 8192 + l * LSTR; };
  auto m1b = [&](int l) { return wb + 8192 + l * LSTR + 16384; };
  auto m2 = [&](int l) { return wb + 8192 + l * LSTR + 32768; };
  auto c1 = [&](int l) { return wb + 8192 + l * LSTR + 49152; };
  auto n1a = [&](int l) { return wb + 8192 + l * LSTR + 65536; };
  auto n1b = [&](int l) { return wb + 8192 + l * LSTR + 81920; };
  auto n2 = [&](int l) { return wb + 8192 + l * LSTR + 98304; };
  __bf16* o1T = wb + 8192 + 4 * LSTR;
  __bf16* o2T = o1T + 16384;

  // ---- shared-memory sizes (dynamic LDS; CDNA5 allows up to 320KB/WG)
  const size_t smemEmbed = 128 * 64 * 2 + 512;
  const size_t smemEdge = 166912;
  const size_t smemNode = 99328 + 32768;
  const size_t smemOut = 66560 + 32768;
  (void)hipFuncSetAttribute((const void*)egnn_edge,
                            hipFuncAttributeMaxDynamicSharedMemorySize, (int)smemEdge);
  (void)hipFuncSetAttribute((const void*)egnn_node,
                            hipFuncAttributeMaxDynamicSharedMemorySize, (int)smemNode);
  (void)hipFuncSetAttribute((const void*)egnn_out,
                            hipFuncAttributeMaxDynamicSharedMemorySize, (int)smemOut);

  // ---- weight prep (bf16, transposed)
  dim3 tb(NTHR);
  int g128 = (128 * HDIM + NTHR - 1) / NTHR;
  int g64 = (64 * HDIM + NTHR - 1) / NTHR;
  prep_transpose<<<g64, tb, 0, stream>>>(embW, embT, 64);
  for (int l = 0; l < L; ++l) {
    prep_transpose<<<g128, tb, 0, stream>>>(msgW1 + (size_t)l * 257 * HDIM, m1a(l), HDIM);
    prep_transpose<<<g128, tb, 0, stream>>>(msgW1 + (size_t)l * 257 * HDIM + 128 * HDIM, m1b(l), HDIM);
    prep_transpose<<<g128, tb, 0, stream>>>(msgW2 + (size_t)l * HDIM * HDIM, m2(l), HDIM);
    prep_transpose<<<g128, tb, 0, stream>>>(coW1 + (size_t)l * HDIM * HDIM, c1(l), HDIM);
    prep_transpose<<<g128, tb, 0, stream>>>(noW1 + (size_t)l * 256 * HDIM, n1a(l), HDIM);
    prep_transpose<<<g128, tb, 0, stream>>>(noW1 + (size_t)l * 256 * HDIM + 128 * HDIM, n1b(l), HDIM);
    prep_transpose<<<g128, tb, 0, stream>>>(noW2 + (size_t)l * HDIM * HDIM, n2(l), HDIM);
  }
  prep_transpose<<<g128, tb, 0, stream>>>(oW1, o1T, HDIM);
  prep_transpose<<<g128, tb, 0, stream>>>(oW2, o2T, HDIM);

  // ---- forward
  int nodeBlocks = (N / 16 + WAVES - 1) / WAVES;  // 391
  int edgeBlocks = 1024;                          // grid-stride over 43125 tiles
  egnn_embed<<<nodeBlocks, NTHR, smemEmbed, stream>>>(h0, posI, embT, embB, h, hbf,
                                                      agg, posc, dpos, N);
  for (int l = 0; l < L; ++l) {
    egnn_edge<<<edgeBlocks, NTHR, smemEdge, stream>>>(
        posc, hbf, ei, E, Etot, m1a(l), m1b(l),
        msgW1 + (size_t)l * 257 * HDIM + 256 * HDIM, msgB1 + l * HDIM, m2(l),
        msgB2 + l * HDIM, c1(l), coB1 + l * HDIM, coW2 + l * HDIM, coB2 + l, agg,
        dpos);
    egnn_node<<<nodeBlocks, NTHR, smemNode, stream>>>(h, hbf, agg, n1a(l), n1b(l),
                                                      noB1 + l * HDIM, n2(l),
                                                      noB2 + l * HDIM, posc, dpos, N);
  }
  egnn_out<<<nodeBlocks, NTHR, smemOut, stream>>>(hbf, o1T, oB1, o2T, oB2, posc,
                                                  (float*)d_out,
                                                  (float*)d_out + (size_t)N * HDIM, N);
}
